// CLA_EP_40613210751435
// MI455X (gfx1250) — matrix operand order repaired and checked by reading, not compiled, not run on hardware
//
#include <hip/hip_runtime.h>
#include <math.h>

// ---------------- problem constants (match reference) ----------------
#define Bb   8
#define Nn   64
#define Dd   768
#define NHh  12
#define DHh  64
#define FFd  1024
#define MNn  (Bb * Nn)          // 512 node rows
#define MEe  (Bb * Nn * Nn)     // 32768 edge rows
#define CATD (3 * Dd)           // 2304

typedef __attribute__((ext_vector_type(16))) _Float16 v16h;
typedef __attribute__((ext_vector_type(8)))  _Float16 h8;
typedef __attribute__((ext_vector_type(8)))  float    v8f;

#define ACT_NONE 0
#define ACT_RELU 1
#define ACT_SILU 2

__device__ __forceinline__ float act_apply(float v, int act) {
  if (act == ACT_RELU) return v > 0.f ? v : 0.f;
  if (act == ACT_SILU) return v / (1.f + __expf(-v));
  return v;
}

// ---------------- WMMA GEMM: C[M,N] = act(A[M,K] @ W[K,N] + bias) (+res) ----
// 256 threads = 8 wave32, 2x4 wave grid, each wave owns a 32x32 sub-tile
// (4 accumulators -> 4 v_wmma_f32_16x16x32_f16 per 32-wide K step).
// Staging: uniform fast path does unguarded global_load_b128 (float4 x2 / x4)
// + packed f16 ds_store_b128; guarded scalar path only for ragged tiles.
#define BM 64
#define BN 128
#define BK 32
#define LDA (BK + 8)    // 40 halves: 16B-aligned vector stores, odd bank stride
#define LDB (BN + 8)    // 136 halves

__global__ void __launch_bounds__(256) k_gemm(
    const float* __restrict__ A, const float* __restrict__ W,
    const float* __restrict__ bias, float* __restrict__ C,
    int M, int N, int K, int ldc, int act, const float* __restrict__ res)
{
  __shared__ _Float16 As[BM][LDA];
  __shared__ _Float16 Bs[BK][LDB];
  const int tid  = threadIdx.x;
  const int wid  = tid >> 5;
  const int lane = tid & 31;
  const int m0 = blockIdx.y * BM;
  const int n0 = blockIdx.x * BN;
  const int wm = (wid >> 2) * 32;    // 0 | 32
  const int wn = (wid & 3) * 32;     // 0..96
  const int mrow = lane & 15;
  const int hi   = lane >> 4;        // 0 | 1 (lane half)
  const bool fullM = (m0 + BM <= M);
  const bool fullN = (n0 + BN <= N);

  v8f acc00 = {}, acc01 = {}, acc10 = {}, acc11 = {};

  for (int k0 = 0; k0 < K; k0 += BK) {
    const bool fullK = (k0 + BK <= K);
    // ---- stage A (64x32) as f16 ----
    if (fullM && fullK) {
      // 8 consecutive floats per thread: 2x global_load_b128, 1x ds_store_b128
      int r = tid >> 2, c = (tid & 3) * 8;
      const float* ap = A + (size_t)(m0 + r) * K + k0 + c;
      float4 u0 = *(const float4*)(ap);
      float4 u1 = *(const float4*)(ap + 4);
      h8 hv;
      hv[0] = (_Float16)u0.x; hv[1] = (_Float16)u0.y;
      hv[2] = (_Float16)u0.z; hv[3] = (_Float16)u0.w;
      hv[4] = (_Float16)u1.x; hv[5] = (_Float16)u1.y;
      hv[6] = (_Float16)u1.z; hv[7] = (_Float16)u1.w;
      *(h8*)&As[r][c] = hv;
    } else {
#pragma unroll
      for (int i = 0; i < (BM * BK) / 256; ++i) {
        int idx = tid + i * 256;
        int r = idx >> 5, c = idx & 31;
        int gm = m0 + r, gk = k0 + c;
        float v = (gm < M && gk < K) ? A[(size_t)gm * K + gk] : 0.f;
        As[r][c] = (_Float16)v;
      }
    }
    // ---- stage B (32x128) as f16 ----
    if (fullN && fullK) {
      // 16 consecutive floats per thread: 4x global_load_b128, 2x ds_store_b128
      int r = tid >> 3, c = (tid & 7) * 16;
      const float* wp = W + (size_t)(k0 + r) * N + n0 + c;
      float4 u0 = *(const float4*)(wp);
      float4 u1 = *(const float4*)(wp + 4);
      float4 u2 = *(const float4*)(wp + 8);
      float4 u3 = *(const float4*)(wp + 12);
      h8 h0, h1;
      h0[0] = (_Float16)u0.x; h0[1] = (_Float16)u0.y;
      h0[2] = (_Float16)u0.z; h0[3] = (_Float16)u0.w;
      h0[4] = (_Float16)u1.x; h0[5] = (_Float16)u1.y;
      h0[6] = (_Float16)u1.z; h0[7] = (_Float16)u1.w;
      h1[0] = (_Float16)u2.x; h1[1] = (_Float16)u2.y;
      h1[2] = (_Float16)u2.z; h1[3] = (_Float16)u2.w;
      h1[4] = (_Float16)u3.x; h1[5] = (_Float16)u3.y;
      h1[6] = (_Float16)u3.z; h1[7] = (_Float16)u3.w;
      *(h8*)&Bs[r][c]     = h0;
      *(h8*)&Bs[r][c + 8] = h1;
    } else {
#pragma unroll
      for (int i = 0; i < (BK * BN) / 256; ++i) {
        int idx = tid + i * 256;
        int r = idx >> 7, c = idx & 127;
        int gk = k0 + r, gn = n0 + c;
        float v = (gk < K && gn < N) ? W[(size_t)gk * N + gn] : 0.f;
        Bs[r][c] = (_Float16)v;
      }
    }
    // prefetch next A tile (global_prefetch_b8)
    if (k0 + 2 * BK <= K && fullM) {
      int r = tid >> 2, c = (tid & 3) * 8;
      __builtin_prefetch(A + (size_t)(m0 + r) * K + k0 + BK + c, 0, 1);
    }
    __syncthreads();

    // A fragments: 16x32 f16 layout (ISA 7.12.2): lane half selects K group
    v16h a0, a1;
#pragma unroll
    for (int j = 0; j < 8; ++j) {
      a0[j]     = As[wm + mrow][hi * 8 + j];
      a0[8 + j] = As[wm + mrow][16 + hi * 8 + j];
      a1[j]     = As[wm + 16 + mrow][hi * 8 + j];
      a1[8 + j] = As[wm + 16 + mrow][16 + hi * 8 + j];
    }
    // B fragments: 32x16, rows striped across VGPR halves, cols across lanes
    v16h b0, b1;
#pragma unroll
    for (int j = 0; j < 16; ++j) {
      b0[j] = Bs[hi * 8 + j + (j & 8)][wn + mrow];
      b1[j] = Bs[hi * 8 + j + (j & 8)][wn + 16 + mrow];
    }
    acc00 = __builtin_amdgcn_wmma_f32_16x16x32_f16(false, a0, false, b0, (short)0, acc00, false, false);
    acc01 = __builtin_amdgcn_wmma_f32_16x16x32_f16(false, a0, false, b1, (short)0, acc01, false, false);
    acc10 = __builtin_amdgcn_wmma_f32_16x16x32_f16(false, a1, false, b0, (short)0, acc10, false, false);
    acc11 = __builtin_amdgcn_wmma_f32_16x16x32_f16(false, a1, false, b1, (short)0, acc11, false, false);
    __syncthreads();
  }

  // epilogue: C/D layout -> VGPR r holds m=r (lanes 0-15) or m=r+8 (16-31), n=lane&15
#pragma unroll
  for (int r = 0; r < 8; ++r) {
    int gm_lo = m0 + wm + (hi ? r + 8 : r);
    int gm_hi = gm_lo + 16;
    int gnA = n0 + wn + mrow;
    int gnB = gnA + 16;
    if (gm_lo < M) {
      if (gnA < N) {
        float v = act_apply(acc00[r] + bias[gnA], act);
        if (res) v += res[(size_t)gm_lo * ldc + gnA];
        C[(size_t)gm_lo * ldc + gnA] = v;
      }
      if (gnB < N) {
        float v = act_apply(acc01[r] + bias[gnB], act);
        if (res) v += res[(size_t)gm_lo * ldc + gnB];
        C[(size_t)gm_lo * ldc + gnB] = v;
      }
    }
    if (gm_hi < M) {
      if (gnA < N) {
        float v = act_apply(acc10[r] + bias[gnA], act);
        if (res) v += res[(size_t)gm_hi * ldc + gnA];
        C[(size_t)gm_hi * ldc + gnA] = v;
      }
      if (gnB < N) {
        float v = act_apply(acc11[r] + bias[gnB], act);
        if (res) v += res[(size_t)gm_hi * ldc + gnB];
        C[(size_t)gm_hi * ldc + gnB] = v;
      }
    }
  }
}

// ---------------- LayerNorm over last dim (Dd), optional fused SiLU --------
__global__ void __launch_bounds__(256) k_layernorm(
    const float* __restrict__ X, const float* __restrict__ g,
    const float* __restrict__ b, float* __restrict__ Y, int do_silu)
{
  __shared__ float s1[256], s2[256];
  size_t row = blockIdx.x;
  const float* x = X + row * Dd;
  float a = 0.f, q = 0.f;
  for (int d = threadIdx.x; d < Dd; d += 256) { float v = x[d]; a += v; q += v * v; }
  s1[threadIdx.x] = a; s2[threadIdx.x] = q; __syncthreads();
  for (int o = 128; o > 0; o >>= 1) {
    if ((int)threadIdx.x < o) { s1[threadIdx.x] += s1[threadIdx.x + o]; s2[threadIdx.x] += s2[threadIdx.x + o]; }
    __syncthreads();
  }
  float mean = s1[0] / Dd;
  float var  = s2[0] / Dd - mean * mean;
  float inv  = rsqrtf(var + 1e-5f);
  for (int d = threadIdx.x; d < Dd; d += 256) {
    float v = (x[d] - mean) * inv * g[d] + b[d];
    if (do_silu) v = v / (1.f + __expf(-v));
    Y[row * Dd + d] = v;
  }
}

// pair = x_i + x_j + e ; out = LN(pair)
__global__ void __launch_bounds__(256) k_pair_ln(
    const float* __restrict__ Xn, const float* __restrict__ E,
    const float* __restrict__ g, const float* __restrict__ b,
    float* __restrict__ Y)
{
  __shared__ float vec[Dd];
  __shared__ float s1[256], s2[256];
  size_t e = blockIdx.x;
  int bi = (int)(e / (Nn * Nn));
  int rem = (int)(e % (Nn * Nn));
  int i = rem / Nn, j = rem % Nn;
  const float* xi = Xn + ((size_t)bi * Nn + i) * Dd;
  const float* xj = Xn + ((size_t)bi * Nn + j) * Dd;
  const float* ee = E + e * Dd;
  float a = 0.f, q = 0.f;
  for (int d = threadIdx.x; d < Dd; d += 256) {
    float v = xi[d] + xj[d] + ee[d];
    vec[d] = v; a += v; q += v * v;
  }
  s1[threadIdx.x] = a; s2[threadIdx.x] = q; __syncthreads();
  for (int o = 128; o > 0; o >>= 1) {
    if ((int)threadIdx.x < o) { s1[threadIdx.x] += s1[threadIdx.x + o]; s2[threadIdx.x] += s2[threadIdx.x + o]; }
    __syncthreads();
  }
  float mean = s1[0] / Dd;
  float inv  = rsqrtf(s2[0] / Dd - mean * mean + 1e-5f);
  for (int d = threadIdx.x; d < Dd; d += 256)
    Y[e * Dd + d] = (vec[d] - mean) * inv * g[d] + b[d];
}

// ---------------- attention: one (b,h) per block, one row per thread -------
__global__ void __launch_bounds__(64) k_attn(
    const float* __restrict__ Q, const float* __restrict__ K,
    const float* __restrict__ V, const int* __restrict__ fmask,
    float* __restrict__ O)
{
  __shared__ float Ks[Nn][DHh + 1];
  __shared__ float Vs[Nn][DHh + 1];
  __shared__ float Ss[Nn][Nn + 1];
  int bh = blockIdx.x;
  int bi = bh / NHh, h = bh % NHh;
  int i = threadIdx.x;
  const size_t base = ((size_t)bi * Nn) * Dd + (size_t)h * DHh;
  for (int d = 0; d < DHh; ++d) {
    Ks[i][d] = K[base + (size_t)i * Dd + d];
    Vs[i][d] = V[base + (size_t)i * Dd + d];
  }
  __syncthreads();
  float mx = -1e30f;
  for (int j = 0; j < Nn; ++j) {
    float dot = 0.f;
    for (int d = 0; d < DHh; ++d) dot += Q[base + (size_t)i * Dd + d] * Ks[j][d];
    dot *= 0.125f;                       // 1/sqrt(64)
    if (fmask[bi * Nn + j] <= 0) dot = -1e9f;
    Ss[i][j] = dot;
    mx = fmaxf(mx, dot);
  }
  float sum = 0.f;
  for (int j = 0; j < Nn; ++j) { float p = __expf(Ss[i][j] - mx); Ss[i][j] = p; sum += p; }
  float inv = 1.f / sum;
  for (int d = 0; d < DHh; ++d) {
    float acc = 0.f;
    for (int j = 0; j < Nn; ++j) acc += Ss[i][j] * Vs[j][d];
    O[base + (size_t)i * Dd + d] = acc * inv;
  }
}

// ---------------- GINE aggregation: out = (1+eps)*xn_i + sum_j relu(xn_j+e)*adj
__global__ void __launch_bounds__(256) k_gine(
    const float* __restrict__ Xn, const float* __restrict__ E,
    const int* __restrict__ em, const float* __restrict__ eps_p,
    float* __restrict__ Out)
{
  int bi = blockIdx.x;                    // (b*Nn + i)
  int b  = bi / Nn;
  float eps1 = 1.f + *eps_p;
  for (int d = threadIdx.x; d < Dd; d += 256) {
    float acc = 0.f;
    for (int j = 0; j < Nn; ++j) {
      if (em[(size_t)bi * Nn + j] != 0) {
        float v = Xn[((size_t)b * Nn + j) * Dd + d] + E[((size_t)bi * Nn + j) * Dd + d];
        acc += v > 0.f ? v : 0.f;
      }
    }
    Out[(size_t)bi * Dd + d] = eps1 * Xn[(size_t)bi * Dd + d] + acc;
  }
}

// ---------------- small elementwise kernels --------------------------------
__global__ void k_time_embed(const int* __restrict__ ts, float* __restrict__ out) {
  int b = blockIdx.x;
  float t = (float)ts[b];
  const int half = Dd / 2;
  for (int d = threadIdx.x; d < Dd; d += blockDim.x) {
    int k = (d < half) ? d : d - half;
    float f = __expf(-logf(10000.f) * (float)k / (float)half);
    float a = t * f;
    out[(size_t)b * Dd + d] = (d < half) ? cosf(a) : sinf(a);
  }
}

__global__ void k_add_class(float* __restrict__ t, const float* __restrict__ ce,
                            const int* __restrict__ lbl) {
  int b = blockIdx.x, c = lbl[b];
  for (int d = threadIdx.x; d < Dd; d += blockDim.x)
    t[(size_t)b * Dd + d] += ce[(size_t)c * Dd + d];
}

__global__ void k_mask_rowmul(const float* __restrict__ X, const int* __restrict__ em,
                              float* __restrict__ Y) {
  size_t e = blockIdx.x;
  float m = em[e] ? 1.f : 0.f;
  for (int d = threadIdx.x; d < Dd; d += blockDim.x) Y[e * Dd + d] = X[e * Dd + d] * m;
}

__global__ void k_pool(const float* __restrict__ EE, float* __restrict__ P) {
  int bi = blockIdx.x;
  for (int d = threadIdx.x; d < Dd; d += blockDim.x) {
    float s = 0.f;
    for (int j = 0; j < Nn; ++j) s += EE[((size_t)bi * Nn + j) * Dd + d];
    P[(size_t)bi * Dd + d] = s * (1.f / Nn);
  }
}

__global__ void k_build_tokens(const float* __restrict__ SE, const float* __restrict__ EE,
                               const float* __restrict__ T, const int* __restrict__ em,
                               float* __restrict__ EZ) {
  size_t e = blockIdx.x;
  int b = (int)(e / (Nn * Nn));
  int rem = (int)(e % (Nn * Nn));
  int i = rem / Nn, j = rem % Nn;
  float m = em[e] ? 1.f : 0.f;
  for (int d = threadIdx.x; d < Dd; d += blockDim.x) {
    float data = (0.5f * (SE[((size_t)b * Nn + i) * Dd + d] + SE[((size_t)b * Nn + j) * Dd + d])
                  + EE[e * Dd + d]) * m;
    EZ[e * Dd + d] = (data + T[(size_t)b * Dd + d]) * m;
  }
}

__global__ void k_node_init(const float* __restrict__ SE, const float* __restrict__ T,
                            const float* __restrict__ P, float* __restrict__ SE2,
                            float* __restrict__ X) {
  int bi = blockIdx.x, b = bi / Nn;
  for (int d = threadIdx.x; d < Dd; d += blockDim.x) {
    float s2 = SE[(size_t)bi * Dd + d] + T[(size_t)b * Dd + d];
    SE2[(size_t)bi * Dd + d] = s2;
    X[(size_t)bi * Dd + d]   = s2 + P[(size_t)bi * Dd + d];
  }
}

__global__ void k_add3(const float* __restrict__ A, const float* __restrict__ B,
                       const float* __restrict__ C, float* __restrict__ Y, int n) {
  int idx = blockIdx.x * blockDim.x + threadIdx.x;
  if (idx < n) Y[idx] = A[idx] + B[idx] + C[idx];
}

__global__ void k_edge_update(float* __restrict__ E, const float* __restrict__ U,
                              const int* __restrict__ em) {
  size_t e = blockIdx.x;
  float m = em[e] ? 1.f : 0.f;
  for (int d = threadIdx.x; d < Dd; d += blockDim.x) E[e * Dd + d] += U[e * Dd + d] * m;
}

__global__ void k_build_cat(const float* __restrict__ EZ, const float* __restrict__ EH,
                            const float* __restrict__ FH, const int* __restrict__ em,
                            float* __restrict__ CT) {
  size_t e = blockIdx.x;
  int b = (int)(e / (Nn * Nn));
  int rem = (int)(e % (Nn * Nn));
  int i = rem / Nn, j = rem % Nn;
  float m = em[e] ? 1.f : 0.f;
  const float* fi = FH + ((size_t)b * Nn + i) * Dd;
  const float* fj = FH + ((size_t)b * Nn + j) * Dd;
  for (int c = threadIdx.x; c < CATD; c += blockDim.x) {
    float v;
    if (c < Dd)           v = EZ[e * Dd + c];
    else if (c < 2 * Dd)  v = EH[e * Dd + (c - Dd)];
    else                  v = fi[c - 2 * Dd] + fj[c - 2 * Dd];
    CT[e * CATD + c] = v * m;
  }
}

__global__ void k_add_ge(const float* __restrict__ EP, const float* __restrict__ GS,
                         const int* __restrict__ em, float* __restrict__ Y) {
  size_t e = blockIdx.x;
  int b = (int)(e / (Nn * Nn));
  int rem = (int)(e % (Nn * Nn));
  int i = rem / Nn, j = rem % Nn;
  float m = em[e] ? 1.f : 0.f;
  for (int d = threadIdx.x; d < Dd; d += blockDim.x)
    Y[e * Dd + d] = EP[e * Dd + d] +
                    (GS[((size_t)b * Nn + i) * Dd + d] + GS[((size_t)b * Nn + j) * Dd + d]) * m;
}

__global__ void k_symmetrize(const float* __restrict__ P, const int* __restrict__ em,
                             float* __restrict__ out) {
  size_t idx = (size_t)blockIdx.x * blockDim.x + threadIdx.x;
  if (idx >= (size_t)MEe * 6) return;
  size_t e = idx / 6;
  int c = (int)(idx % 6);
  int b = (int)(e / (Nn * Nn));
  int rem = (int)(e % (Nn * Nn));
  int i = rem / Nn, j = rem % Nn;
  size_t et = ((size_t)b * Nn + j) * Nn + i;
  float a  = P[e * 6 + c]  * (em[e]  ? 1.f : 0.f);
  float a2 = P[et * 6 + c] * (em[et] ? 1.f : 0.f);
  out[idx] = 0.5f * (a + a2);
}

// ---------------- host-side helpers ----------------------------------------
static inline void gemm(hipStream_t s, const float* A, const float* W, const float* b,
                        float* C, int M, int N, int K, int ldc, int act, const float* res) {
  dim3 grid((N + BN - 1) / BN, (M + BM - 1) / BM);
  k_gemm<<<grid, 256, 0, s>>>(A, W, b, C, M, N, K, ldc, act, res);
}
static inline void lnorm(hipStream_t s, const float* X, const float* g, const float* b,
                         float* Y, int rows, int silu) {
  k_layernorm<<<rows, 256, 0, s>>>(X, g, b, Y, silu);
}

struct Lin { const float *b, *w; };
struct Mlp { const float *be, *g, *l1b, *l1w, *l2b, *l2w; };
struct TfL { Lin ak, ao, aq, av; const float *b1, *b2; Lin f1, f2; const float *g1, *g2; };
struct Gnn {
  Lin ak, ao, aq, av;
  const float *bee, *bn1, *bn2;
  Lin e1, e2;
  const float* eps;
  Lin f1, f2;
  const float* ge;
  Lin gine1, gine2;
  const float *gn1, *gn2;
  Lin lin_e, lin_f;
};

extern "C" void kernel_launch(void* const* d_in, const int* in_sizes, int n_in,
                              void* d_out, int out_size, void* d_ws, size_t ws_size,
                              hipStream_t stream) {
  (void)in_sizes; (void)n_in; (void)out_size; (void)ws_size;
  const float* edgePos = (const float*)d_in[0];   // [B,N,N,6]
  const float* surfPos = (const float*)d_in[1];   // [B,N,6]
  const float* surfZ   = (const float*)d_in[2];   // [B,N,48]
  const float* prm     = (const float*)d_in[3];   // pytree leaves, sorted-key order
  const int* timesteps = (const int*)d_in[4];     // [B]
  const int* face_mask = (const int*)d_in[5];     // [B,N]
  const int* edge_mask = (const int*)d_in[6];     // [B,N,N,1] -> flat [ME]
  const int* class_lbl = (const int*)d_in[7];     // [B]
  float* outp = (float*)d_out;

  // ---- parameter offsets: JAX tree_leaves order (dict keys sorted) ----
  size_t off = 0;
  auto take = [&](size_t n) { const float* r = prm + off; off += n; return r; };
  auto takeLin = [&](int din, int dout) { Lin l; l.b = take(dout); l.w = take((size_t)din * dout); return l; };
  auto takeMlp = [&](int din, int dh, int dout) {
    Mlp m; m.be = take(dh); m.g = take(dh); m.l1b = take(dh);
    m.l1w = take((size_t)din * dh); m.l2b = take(dout); m.l2w = take((size_t)dh * dout); return m;
  };
  const float* class_embed = take((size_t)11 * Dd);
  Mlp edgep   = takeMlp(6, Dd, Dd);
  Mlp fc_out  = takeMlp(Dd, Dd, 6);
  Mlp fc_out2 = takeMlp(CATD, Dd, Dd);
  Mlp fc_pool = takeMlp(Dd, Dd, Dd);
  Gnn gnn[6];
  for (int l = 0; l < 6; ++l) {
    Gnn& g = gnn[l];
    // block: attn{k,o,q,v}, bee, bn1, bn2, e1, e2, eps, f1, f2, ge, gine1, gine2, gn1, gn2
    g.ak = takeLin(Dd, Dd); g.ao = takeLin(Dd, Dd); g.aq = takeLin(Dd, Dd); g.av = takeLin(Dd, Dd);
    g.bee = take(Dd); g.bn1 = take(Dd); g.bn2 = take(Dd);
    g.e1 = takeLin(Dd, Dd); g.e2 = takeLin(Dd, Dd);
    g.eps = take(1);
    g.f1 = takeLin(Dd, FFd); g.f2 = takeLin(FFd, Dd);
    g.ge = take(Dd);
    g.gine1 = takeLin(Dd, Dd); g.gine2 = takeLin(Dd, Dd);
    g.gn1 = take(Dd); g.gn2 = take(Dd);
    g.lin_e = takeLin(Dd, 128); g.lin_f = takeLin(Dd, 128);
  }
  Mlp surfp = takeMlp(6, Dd, Dd);
  Mlp surfz = takeMlp(48, Dd, Dd);
  TfL tf[12];
  for (int l = 0; l < 12; ++l) {
    TfL& t = tf[l];
    t.ak = takeLin(Dd, Dd); t.ao = takeLin(Dd, Dd); t.aq = takeLin(Dd, Dd); t.av = takeLin(Dd, Dd);
    t.b1 = take(Dd); t.b2 = take(Dd);
    t.f1 = takeLin(Dd, FFd); t.f2 = takeLin(FFd, Dd);
    t.g1 = take(Dd); t.g2 = take(Dd);
  }
  const float* tfn_b = take(Dd);
  const float* tfn_g = take(Dd);
  Mlp timeM = takeMlp(Dd, Dd, Dd);

  // ---- workspace layout (floats) ----
  float* ws = (float*)d_ws;
  size_t wo = 0;
  auto walloc = [&](size_t n) { float* r = ws + wo; wo += n; return r; };
  const size_t ED = (size_t)MEe * Dd;
  float* ez   = walloc(ED);            // dense_ez (tokens)
  float* eh   = walloc(ED);            // concat of lin_e heads
  float* t0   = walloc(ED);
  float* t1   = walloc(ED);
  float* t2   = walloc(ED);
  float* ebuf = walloc(ED);            // running edge features e
  float* cat  = ebuf;                  // [ME,2304] overlays ebuf (+2 slabs) once e is dead
  walloc(ED * 2);                      // tail of cat
  float* se   = walloc((size_t)MNn * Dd);
  float* se2  = walloc((size_t)MNn * Dd);   // GNN node state h (in-place)
  float* x    = walloc((size_t)MNn * Dd);   // TF node state
  float* xn   = walloc((size_t)MNn * Dd);
  float* pool = walloc((size_t)MNn * Dd);
  float* fh   = walloc((size_t)MNn * Dd);   // concat of lin_f heads
  float* gs   = walloc((size_t)MNn * Dd);   // global_surf
  float* nq   = walloc((size_t)MNn * Dd);
  float* nk   = walloc((size_t)MNn * Dd);
  float* nv   = walloc((size_t)MNn * Dd);
  float* natt = walloc((size_t)MNn * Dd);
  float* nff  = walloc((size_t)MNn * FFd);
  float* na   = walloc((size_t)MNn * FFd);
  float* nb   = walloc((size_t)MNn * FFd);
  float* tvec = walloc((size_t)Bb * Dd);
  float* tsc  = walloc((size_t)Bb * Dd);
  float* t8a  = walloc((size_t)Bb * Dd);
  float* t8b  = walloc((size_t)Bb * Dd);

  // ---- time / class embedding ----
  k_time_embed<<<Bb, 256, 0, stream>>>(timesteps, tsc);
  gemm(stream, tsc, timeM.l1w, timeM.l1b, t8a, Bb, Dd, Dd, Dd, ACT_NONE, nullptr);
  lnorm(stream, t8a, timeM.g, timeM.be, t8b, Bb, 1);
  gemm(stream, t8b, timeM.l2w, timeM.l2b, tvec, Bb, Dd, Dd, Dd, ACT_NONE, nullptr);
  k_add_class<<<Bb, 256, 0, stream>>>(tvec, class_embed, class_lbl);

  // ---- surface embeddings: se = surfp(surfPos) + surfz(surfZ) ----
  gemm(stream, surfPos, surfp.l1w, surfp.l1b, na, MNn, Dd, 6, Dd, ACT_NONE, nullptr);
  lnorm(stream, na, surfp.g, surfp.be, nb, MNn, 1);
  gemm(stream, nb, surfp.l2w, surfp.l2b, se, MNn, Dd, Dd, Dd, ACT_NONE, nullptr);
  gemm(stream, surfZ, surfz.l1w, surfz.l1b, na, MNn, Dd, 48, Dd, ACT_NONE, nullptr);
  lnorm(stream, na, surfz.g, surfz.be, nb, MNn, 1);
  gemm(stream, nb, surfz.l2w, surfz.l2b, se, MNn, Dd, Dd, Dd, ACT_NONE, se);  // se += ...

  // ---- edge embedding: ee = edgep(edgePos) * em ----
  gemm(stream, edgePos, edgep.l1w, edgep.l1b, t0, MEe, Dd, 6, Dd, ACT_NONE, nullptr);
  lnorm(stream, t0, edgep.g, edgep.be, t1, MEe, 1);
  gemm(stream, t1, edgep.l2w, edgep.l2b, t0, MEe, Dd, Dd, Dd, ACT_NONE, nullptr);
  k_mask_rowmul<<<MEe, 256, 0, stream>>>(t0, edge_mask, t1);        // t1 = ee
  k_pool<<<MNn, 256, 0, stream>>>(t1, pool);                        // pooled = mean_j ee
  k_build_tokens<<<MEe, 256, 0, stream>>>(se, t1, tvec, edge_mask, ez);
  k_node_init<<<MNn, 256, 0, stream>>>(se, tvec, pool, se2, x);     // se2=se+t ; x=se+t+pool
  hipMemcpyAsync(ebuf, ez, ED * sizeof(float), hipMemcpyDeviceToDevice, stream);

  // ---- 12 transformer layers on x ----
  for (int l = 0; l < 12; ++l) {
    const TfL& t = tf[l];
    lnorm(stream, x, t.g1, t.b1, xn, MNn, 0);
    gemm(stream, xn, t.aq.w, t.aq.b, nq, MNn, Dd, Dd, Dd, ACT_NONE, nullptr);
    gemm(stream, xn, t.ak.w, t.ak.b, nk, MNn, Dd, Dd, Dd, ACT_NONE, nullptr);
    gemm(stream, xn, t.av.w, t.av.b, nv, MNn, Dd, Dd, Dd, ACT_NONE, nullptr);
    k_attn<<<Bb * NHh, 64, 0, stream>>>(nq, nk, nv, face_mask, natt);
    gemm(stream, natt, t.ao.w, t.ao.b, x, MNn, Dd, Dd, Dd, ACT_NONE, x);      // x += attn
    lnorm(stream, x, t.g2, t.b2, xn, MNn, 0);
    gemm(stream, xn, t.f1.w, t.f1.b, nff, MNn, FFd, Dd, FFd, ACT_RELU, nullptr);
    gemm(stream, nff, t.f2.w, t.f2.b, x, MNn, Dd, FFd, Dd, ACT_NONE, x);      // x += ffn
  }
  lnorm(stream, x, tfn_g, tfn_b, gs, MNn, 0);                                  // global_surf

  // ---- 6 LTIA GNN layers on (se2, ebuf) ----
  for (int l = 0; l < 6; ++l) {
    const Gnn& g = gnn[l];
    lnorm(stream, se2, g.gn1, g.bn1, xn, MNn, 0);
    k_gine<<<MNn, 256, 0, stream>>>(xn, ebuf, edge_mask, g.eps, na);           // (1+eps)xn + agg
    gemm(stream, na, g.gine1.w, g.gine1.b, nb, MNn, Dd, Dd, Dd, ACT_SILU, nullptr);
    gemm(stream, nb, g.gine2.w, g.gine2.b, na, MNn, Dd, Dd, Dd, ACT_NONE, nullptr);  // local
    gemm(stream, xn, g.aq.w, g.aq.b, nq, MNn, Dd, Dd, Dd, ACT_NONE, nullptr);
    gemm(stream, xn, g.ak.w, g.ak.b, nk, MNn, Dd, Dd, Dd, ACT_NONE, nullptr);
    gemm(stream, xn, g.av.w, g.av.b, nv, MNn, Dd, Dd, Dd, ACT_NONE, nullptr);
    k_attn<<<Bb * NHh, 64, 0, stream>>>(nq, nk, nv, face_mask, natt);
    gemm(stream, natt, g.ao.w, g.ao.b, nb, MNn, Dd, Dd, Dd, ACT_NONE, nullptr); // glob
    k_add3<<<(MNn * Dd + 255) / 256, 256, 0, stream>>>(se2, na, nb, se2, MNn * Dd);
    lnorm(stream, se2, g.gn2, g.bn2, xn, MNn, 0);
    gemm(stream, xn, g.f1.w, g.f1.b, nff, MNn, FFd, Dd, FFd, ACT_SILU, nullptr);
    gemm(stream, nff, g.f2.w, g.f2.b, se2, MNn, Dd, FFd, Dd, ACT_NONE, se2);   // h += ffn
    // edge update
    k_pair_ln<<<MEe, 256, 0, stream>>>(se2, ebuf, g.ge, g.bee, t0);
    gemm(stream, t0, g.e1.w, g.e1.b, t1, MEe, Dd, Dd, Dd, ACT_SILU, nullptr);
    gemm(stream, t1, g.e2.w, g.e2.b, t2, MEe, Dd, Dd, Dd, ACT_NONE, nullptr);
    k_edge_update<<<MEe, 256, 0, stream>>>(ebuf, t2, edge_mask);               // e += u*em
    // hidden heads written straight into their concat slots (ldc=768)
    gemm(stream, se2,  g.lin_f.w, g.lin_f.b, fh + l * 128, MNn, 128, Dd, Dd, ACT_NONE, nullptr);
    gemm(stream, ebuf, g.lin_e.w, g.lin_e.b, eh + l * 128, MEe, 128, Dd, Dd, ACT_NONE, nullptr);
  }

  // ---- output head ----
  k_build_cat<<<MEe, 256, 0, stream>>>(ez, eh, fh, edge_mask, cat);   // ebuf dead -> reused
  gemm(stream, cat, fc_out2.l1w, fc_out2.l1b, t0, MEe, Dd, CATD, Dd, ACT_NONE, nullptr);
  lnorm(stream, t0, fc_out2.g, fc_out2.be, t1, MEe, 1);
  gemm(stream, t1, fc_out2.l2w, fc_out2.l2b, t2, MEe, Dd, Dd, Dd, ACT_NONE, nullptr); // ep
  k_add_ge<<<MEe, 256, 0, stream>>>(t2, gs, edge_mask, t0);                            // ep+ge
  gemm(stream, t0, fc_pool.l1w, fc_pool.l1b, t1, MEe, Dd, Dd, Dd, ACT_NONE, nullptr);
  lnorm(stream, t1, fc_pool.g, fc_pool.be, t2, MEe, 1);
  gemm(stream, t2, fc_pool.l2w, fc_pool.l2b, t0, MEe, Dd, Dd, Dd, ACT_NONE, nullptr);
  gemm(stream, t0, fc_out.l1w, fc_out.l1b, t1, MEe, Dd, Dd, Dd, ACT_NONE, nullptr);
  lnorm(stream, t1, fc_out.g, fc_out.be, t2, MEe, 1);
  gemm(stream, t2, fc_out.l2w, fc_out.l2b, t1, MEe, 6, Dd, 6, ACT_NONE, nullptr);     // [ME,6]
  k_symmetrize<<<((size_t)MEe * 6 + 255) / 256, 256, 0, stream>>>(t1, edge_mask, outp);
}
